// ChannelDense_69045894250869
// MI455X (gfx1250) — matrix operand order repaired and checked by reading, not compiled, not run on hardware
//
#include <hip/hip_runtime.h>

typedef __attribute__((ext_vector_type(16))) _Float16 v16h;
typedef __attribute__((ext_vector_type(8)))  float    v8f;

#define BATCH 128
#define NPTS  2048
#define CDIM  256   // IN == OUT == 256
#define NCHAN 64

union Frag16 { v16h v; uint4 u[2]; };
union Pack4  { _Float16 h[4]; uint2 u; };

// ---------------------------------------------------------------------------
// Prepass: convert weights fp32 -> f16 into workspace (64*256*256 = 4.19M elem)
// ---------------------------------------------------------------------------
__global__ __launch_bounds__(256) void cd_cvt_w(const float* __restrict__ w,
                                                _Float16* __restrict__ w16,
                                                int nelem) {
    int i = (blockIdx.x * 256 + threadIdx.x) * 4;
    if (i < nelem) {
        float4 f = *(const float4*)(w + i);
        Pack4 p;
        p.h[0] = (_Float16)f.x; p.h[1] = (_Float16)f.y;
        p.h[2] = (_Float16)f.z; p.h[3] = (_Float16)f.w;
        *(uint2*)(w16 + i) = p.u;
    }
}

// ---------------------------------------------------------------------------
// Main kernel: one workgroup per point n. 512 threads = 16 waves.
//   wave = 2*(row strip of 16) + (column half of 128)
//   each wave: 8 accumulator tiles 16x16 (f32), K-loop step 32 via WMMA f16.
// ---------------------------------------------------------------------------
__global__ __launch_bounds__(512) void cd_main(const float* __restrict__ x,
                                               const int* __restrict__ channels,
                                               const _Float16* __restrict__ w16,
                                               const float* __restrict__ bias,
                                               float* __restrict__ out) {
    __shared__ _Float16 xs[BATCH * CDIM];   // 64 KB: x[:, n, :] as f16

    const int n = blockIdx.x;
    const int c = channels[n];
    const int tid = threadIdx.x;

    // ---- Phase 1: stage x[:, n, :] into LDS (fp32 -> f16). 4 threads/row. ----
    {
        const int r = tid >> 2;       // batch row 0..127
        const int q = tid & 3;        // quarter of 64 floats
        const float* src = x + ((size_t)r * NPTS + n) * CDIM + q * 64;
        _Float16*    dst = xs + r * CDIM + q * 64;
#pragma unroll
        for (int j = 0; j < 64; j += 4) {
            float4 f = *(const float4*)(src + j);
            Pack4 p;
            p.h[0] = (_Float16)f.x; p.h[1] = (_Float16)f.y;
            p.h[2] = (_Float16)f.z; p.h[3] = (_Float16)f.w;
            *(uint2*)(dst + j) = p.u;
        }
    }
    __syncthreads();

    // ---- Phase 2: WMMA K-loop ----
    const int wave = tid >> 5;        // 0..15
    const int lane = tid & 31;
    const int ms   = wave >> 1;       // row strip (16 rows each)
    const int chf  = wave & 1;        // column half (128 cols each)
    const int ln   = lane & 15;
    const int hl   = lane >> 4;       // half-wave id

    v8f acc[8] = {};                  // 8 tiles x 8 f32 = 64 VGPRs

    const int arow = ms * 16 + ln;
    const _Float16* abase = xs + arow * CDIM;
    // B: element (K=i, N=o) = W[c][o][i]; lane holds fixed o, contiguous i.
    const _Float16* wbase = w16 + ((size_t)c * CDIM + chf * 128 + ln) * CDIM;

    for (int k0 = 0; k0 < CDIM; k0 += 32) {
        // A fragment: lanes 0-15 hold K = k0+{0..7, 16..23}; lanes 16-31 +8.
        Frag16 a;
        const _Float16* ap = abase + k0 + hl * 8;
        a.u[0] = *(const uint4*)(ap);
        a.u[1] = *(const uint4*)(ap + 16);
#pragma unroll
        for (int t = 0; t < 8; ++t) {
            // B fragment: lanes 0-15 hold K = k0+{0..7, 16..23}; lanes 16-31 +8.
            Frag16 b;
            const _Float16* bp = wbase + (size_t)t * 16 * CDIM + k0 + hl * 8;
            b.u[0] = *(const uint4*)(bp);
            b.u[1] = *(const uint4*)(bp + 16);
            acc[t] = __builtin_amdgcn_wmma_f32_16x16x32_f16(
                false, a.v, false, b.v, (short)0, acc[t], false, false);
        }
    }

    // ---- Phase 3: bias + tanh + residual + store ----
    const int col0 = chf * 128 + ln;
#pragma unroll
    for (int t = 0; t < 8; ++t) {
        const int col = col0 + t * 16;
        const float bv = bias[c * CDIM + col];
#pragma unroll
        for (int r = 0; r < 8; ++r) {
            const int row = ms * 16 + r + hl * 8;   // C/D layout: lanes16-31 -> M+8
            float v = acc[t][r] + bv;
            v = tanhf(v);
            v += (float)xs[row * CDIM + col];       // residual (x as f16)
            out[((size_t)row * NPTS + n) * CDIM + col] = v;
        }
    }
}

// ---------------------------------------------------------------------------
// Tail: pass-through `channels` output appended after y in d_out.
// tail == NPTS   -> write as float values
// tail == 2*NPTS -> write as raw little-endian int64 (low = ch, high = 0)
// ---------------------------------------------------------------------------
__global__ __launch_bounds__(256) void cd_tail(const int* __restrict__ ch,
                                               float* __restrict__ tail_out,
                                               int tail) {
    int i = blockIdx.x * 256 + threadIdx.x;
    if (i >= NPTS) return;
    if (tail == NPTS) {
        tail_out[i] = (float)ch[i];
    } else if (tail == 2 * NPTS) {
        int2 v; v.x = ch[i]; v.y = 0;
        ((int2*)tail_out)[i] = v;
    }
}

extern "C" void kernel_launch(void* const* d_in, const int* in_sizes, int n_in,
                              void* d_out, int out_size, void* d_ws, size_t ws_size,
                              hipStream_t stream) {
    const float* x        = (const float*)d_in[0];
    const int*   channels = (const int*)d_in[1];
    const float* weight   = (const float*)d_in[2];
    const float* bias     = (const float*)d_in[3];
    float*       out      = (float*)d_out;

    _Float16* w16 = (_Float16*)d_ws;              // 64*256*256 f16 = 8.39 MB
    const int wElem = NCHAN * CDIM * CDIM;        // 4,194,304

    // 1) convert weights to f16 (L2-resident for the GEMM phase)
    cd_cvt_w<<<wElem / 4 / 256, 256, 0, stream>>>(weight, w16, wElem);

    // 2) main fused gather-GEMM + bias + tanh + residual
    cd_main<<<NPTS, 512, 0, stream>>>(x, channels, w16, bias, out);

    // 3) channels pass-through output
    const long long ySize = (long long)BATCH * NPTS * CDIM;
    const int tail = (int)((long long)out_size - ySize);
    if (tail > 0) {
        cd_tail<<<(NPTS + 255) / 256, 256, 0, stream>>>(channels, out + ySize, tail);
    }
}
